// GAE_REGRESSION_41188736369293
// MI455X (gfx1250) — matrix operand order repaired and checked by reading, not compiled, not run on hardware
//
#include <hip/hip_runtime.h>

#define N_NODES 100000
#define N_EDGES 3200000
#define F_IN 512
#define H1 256
#define H2 128

typedef __attribute__((ext_vector_type(16))) __bf16 v16bf;
typedef __attribute__((ext_vector_type(8)))  float  v8f;
typedef __attribute__((ext_vector_type(16))) unsigned short v16us;
typedef __attribute__((ext_vector_type(8)))  unsigned short v8us;

__device__ __forceinline__ unsigned short f2bf(float f) {
  unsigned int u = __float_as_uint(f);
  u += 0x7FFFu + ((u >> 16) & 1u);   // round-to-nearest-even
  return (unsigned short)(u >> 16);
}

// ---------------------------------------------------------------------------
// Streaming f32 -> bf16 conversion, 8 elements per thread (16B stores).
// ---------------------------------------------------------------------------
__global__ void f32_to_bf16(const float* __restrict__ in, unsigned short* __restrict__ out,
                            long long n8) {
  long long i = (long long)blockIdx.x * blockDim.x + threadIdx.x;
  if (i >= n8) return;
  const float4* p = (const float4*)(in + i * 8);
  float4 a = p[0];
  float4 b = p[1];
  v8us o;
  o[0] = f2bf(a.x); o[1] = f2bf(a.y); o[2] = f2bf(a.z); o[3] = f2bf(a.w);
  o[4] = f2bf(b.x); o[5] = f2bf(b.y); o[6] = f2bf(b.z); o[7] = f2bf(b.w);
  *(v8us*)(out + i * 8) = o;
}

// ---------------------------------------------------------------------------
// Pack W[K,N] (f32 row-major) -> out[N,K] bf16 (B^T) so WMMA B-fragment loads
// are contiguous K-runs per output column.
// ---------------------------------------------------------------------------
__global__ void pack_wt(const float* __restrict__ W, unsigned short* __restrict__ out,
                        int K, int N, int log2N) {
  int idx = blockIdx.x * blockDim.x + threadIdx.x;
  if (idx >= K * N) return;
  int k = idx >> log2N;
  int n = idx & (N - 1);
  out[(size_t)n * K + k] = f2bf(W[idx]);
}

// ---------------------------------------------------------------------------
// WMMA fragment loaders (bf16, 16x16x32 shape).
// A 16x32: lane<16 row M=lane, elems 0..7 -> K=k0..k0+7, elems 8..15 ->
//          K=k0+16..k0+23; lane>=16 rows M=lane-16, +8 on both K runs.
// B 32x16: lane L col N=L&15; same K layout as A (elems 0..7 -> K=k0+8*hi..,
//          elems 8..15 -> K=k0+16+8*hi..).
// ---------------------------------------------------------------------------
__device__ __forceinline__ v16us load_frag_a(const unsigned short* __restrict__ row,
                                             int k0, int hi) {
  int ca = k0 + hi * 8;
  v8us a0 = *(const v8us*)(row + ca);        // K = ca .. ca+7
  v8us a1 = *(const v8us*)(row + ca + 16);   // K = ca+16 .. ca+23
  return __builtin_shufflevector(a0, a1, 0,1,2,3,4,5,6,7,8,9,10,11,12,13,14,15);
}

__device__ __forceinline__ v16us load_frag_b(const unsigned short* __restrict__ row,
                                             int k0, int hi) {
  int cb = k0 + hi * 8;
  v8us b0 = *(const v8us*)(row + cb);
  v8us b1 = *(const v8us*)(row + cb + 16);
  return __builtin_shufflevector(b0, b1, 0,1,2,3,4,5,6,7,8,9,10,11,12,13,14,15);
}

__device__ __forceinline__ v8f wmma_bf16(v16us a, v16us b, v8f c) {
  return __builtin_amdgcn_wmma_f32_16x16x32_bf16(
      false, __builtin_bit_cast(v16bf, a),
      false, __builtin_bit_cast(v16bf, b),
      (short)0, c, false, false);
}

// C/D layout: lane<16 N=lane, VGPR r -> M=r ; lane>=16 N=lane-16, VGPR r -> M=8+r.
__device__ __forceinline__ void store_tile(float* __restrict__ C, int N, int mb, int n,
                                           v8f acc) {
  #pragma unroll
  for (int r = 0; r < 8; ++r)
    C[(size_t)(mb + r) * N + n] = acc[r];
}

// ---------------------------------------------------------------------------
// GEMM C[M,N] = A[M,K] (bf16) x Bp[N,K] (bf16, pre-transposed).
// One wave computes a 32x32 output block: 2 A-fragments x 2 B-fragments,
// 4 accumulators, 4 v_wmma per k-step -> 2x fragment reuse vs 16x16 tiling.
// ---------------------------------------------------------------------------
__global__ __launch_bounds__(256) void gemm_bf16_2x2(
    const unsigned short* __restrict__ A, const unsigned short* __restrict__ Bp,
    float* __restrict__ C, int M, int N, int K) {
  int lane = threadIdx.x & 31;
  int wave = (blockIdx.x * blockDim.x + threadIdx.x) >> 5;
  int nt = N >> 5;                       // 32-wide output tiles
  int tm = wave / nt;
  int tn = wave - tm * nt;
  if (tm * 32 >= M) return;              // wave-uniform guard (EXEC stays full)
  int l15 = lane & 15;
  int hi  = lane >> 4;
  const unsigned short* a0row = A + (size_t)(tm * 32 + l15) * K;
  const unsigned short* a1row = a0row + (size_t)16 * K;
  const unsigned short* b0row = Bp + (size_t)(tn * 32 + l15) * K;
  const unsigned short* b1row = b0row + (size_t)16 * K;
  v8f c00 = {}, c01 = {}, c10 = {}, c11 = {};
  for (int k0 = 0; k0 < K; k0 += 32) {
    v16us a0 = load_frag_a(a0row, k0, hi);
    v16us a1 = load_frag_a(a1row, k0, hi);
    v16us b0 = load_frag_b(b0row, k0, hi);
    v16us b1 = load_frag_b(b1row, k0, hi);
    c00 = wmma_bf16(a0, b0, c00);
    c01 = wmma_bf16(a0, b1, c01);
    c10 = wmma_bf16(a1, b0, c10);
    c11 = wmma_bf16(a1, b1, c11);
  }
  int n0  = tn * 32 + l15;
  int mb0 = tm * 32 + hi * 8;
  store_tile(C, N, mb0,      n0,      c00);
  store_tile(C, N, mb0,      n0 + 16, c01);
  store_tile(C, N, mb0 + 16, n0,      c10);
  store_tile(C, N, mb0 + 16, n0 + 16, c11);
}

// ---------------------------------------------------------------------------
// Edge-parallel SpMM: agg[dst] += ew * S[src]; one thread per (edge, 4-feature
// chunk). S and agg both fit in the 192MB L2, so atomics stay on-chip.
// ---------------------------------------------------------------------------
__global__ void spmm_atomic(const float* __restrict__ S, const int* __restrict__ src,
                            const int* __restrict__ dst, const float* __restrict__ ew,
                            float* __restrict__ agg, int H, int log2c, long long total) {
  long long idx = (long long)blockIdx.x * blockDim.x + threadIdx.x;
  if (idx >= total) return;
  int e = (int)(idx >> log2c);
  int c = ((int)idx) & ((1 << log2c) - 1);
  int s = src[e];
  int d = dst[e];
  float w = ew[e];
  float4 v = *(const float4*)(S + (size_t)s * H + c * 4);
  float* p = agg + (size_t)d * H + c * 4;
  unsafeAtomicAdd(p + 0, v.x * w);
  unsafeAtomicAdd(p + 1, v.y * w);
  unsafeAtomicAdd(p + 2, v.z * w);
  unsafeAtomicAdd(p + 3, v.w * w);
}

// Fused BatchNorm(eval) + ReLU, emitting bf16 for the next WMMA GEMM.
__global__ void bn_relu_to_bf16(const float* __restrict__ agg, const float* __restrict__ g,
                                const float* __restrict__ b, const float* __restrict__ m,
                                const float* __restrict__ v, unsigned short* __restrict__ out,
                                int maskH, long long total) {
  long long idx = (long long)blockIdx.x * blockDim.x + threadIdx.x;
  if (idx >= total) return;
  int j = ((int)idx) & maskH;
  float sc = g[j] * rsqrtf(v[j] + 1e-5f);
  float x = (agg[idx] - m[j]) * sc + b[j];
  out[idx] = f2bf(fmaxf(x, 0.0f));
}

// One wave per node: BN2 (no activation) -> mu; pred = dot(mu, linW) + linb.
// Writes pred [100000], mu [100000x128], logvar (== mu) [100000x128].
__global__ void finalize(const float* __restrict__ agg2, const float* __restrict__ g,
                         const float* __restrict__ b, const float* __restrict__ m,
                         const float* __restrict__ v, const float* __restrict__ linW,
                         const float* __restrict__ linb, float* __restrict__ out,
                         int nNodes) {
  int wave = (blockIdx.x * blockDim.x + threadIdx.x) >> 5;
  int lane = threadIdx.x & 31;
  if (wave >= nNodes) return;
  const float* row = agg2 + (size_t)wave * H2;
  float partial = 0.0f;
  float mu[4];
  #pragma unroll
  for (int t = 0; t < 4; ++t) {
    int j = t * 32 + lane;
    float sc = g[j] * rsqrtf(v[j] + 1e-5f);
    float mm = (row[j] - m[j]) * sc + b[j];
    mu[t] = mm;
    partial += mm * linW[j];
  }
  #pragma unroll
  for (int off = 16; off > 0; off >>= 1)
    partial += __shfl_down(partial, off);
  float* muOut = out + N_NODES + (size_t)wave * H2;
  float* lvOut = out + N_NODES + (size_t)N_NODES * H2 + (size_t)wave * H2;
  #pragma unroll
  for (int t = 0; t < 4; ++t) {
    int j = t * 32 + lane;
    muOut[j] = mu[t];
    lvOut[j] = mu[t];
  }
  if (lane == 0) out[wave] = partial + linb[0];
}

extern "C" void kernel_launch(void* const* d_in, const int* in_sizes, int n_in,
                              void* d_out, int out_size, void* d_ws, size_t ws_size,
                              hipStream_t stream) {
  (void)in_sizes; (void)n_in; (void)out_size; (void)ws_size;
  const float* x    = (const float*)d_in[0];
  const int*   esrc = (const int*)  d_in[1];
  const int*   edst = (const int*)  d_in[2];
  const float* ew   = (const float*)d_in[3];
  const float* W1   = (const float*)d_in[4];
  const float* g1   = (const float*)d_in[5];
  const float* b1   = (const float*)d_in[6];
  const float* m1   = (const float*)d_in[7];
  const float* v1   = (const float*)d_in[8];
  const float* W2   = (const float*)d_in[9];
  const float* g2   = (const float*)d_in[10];
  const float* b2   = (const float*)d_in[11];
  const float* m2   = (const float*)d_in[12];
  const float* v2   = (const float*)d_in[13];
  const float* linW = (const float*)d_in[14];
  const float* linb = (const float*)d_in[15];
  float* out = (float*)d_out;

  char* ws = (char*)d_ws;
  size_t off = 0;
  auto alloc = [&](size_t bytes) -> char* {
    char* p = ws + off;
    off += (bytes + 255) & ~(size_t)255;
    return p;
  };
  // xb is dead after GEMM1; support2 + agg2 alias its storage (stream-ordered).
  char*           xb_region = alloc((size_t)N_NODES * F_IN * 2);   // 102.4 MB
  unsigned short* xb        = (unsigned short*)xb_region;
  float*          support2  = (float*)xb_region;                            // 51.2 MB
  float*          agg2      = (float*)(xb_region + (size_t)N_NODES * H2 * 4); // 51.2 MB
  float*          support1  = (float*)         alloc((size_t)N_NODES * H1 * 4);
  float*          agg1      = (float*)         alloc((size_t)N_NODES * H1 * 4);
  unsigned short* h1b       = (unsigned short*)alloc((size_t)N_NODES * H1 * 2);
  unsigned short* w1p       = (unsigned short*)alloc((size_t)F_IN * H1 * 2);
  unsigned short* w2p       = (unsigned short*)alloc((size_t)H1 * H2 * 2);

  // Prep: pack weights, convert x to bf16, zero layer-1 scatter target.
  pack_wt<<<(F_IN * H1 + 255) / 256, 256, 0, stream>>>(W1, w1p, F_IN, H1, 8);
  pack_wt<<<(H1 * H2 + 255) / 256, 256, 0, stream>>>(W2, w2p, H1, H2, 7);
  long long n8 = (long long)N_NODES * F_IN / 8;
  f32_to_bf16<<<(unsigned)((n8 + 255) / 256), 256, 0, stream>>>(x, xb, n8);
  hipMemsetAsync(agg1, 0, (size_t)N_NODES * H1 * 4, stream);

  // Layer 1: WMMA GEMM -> SpMM (L2-resident atomics) -> BN+ReLU (bf16 out)
  {
    int waves = (N_NODES / 32) * (H1 / 32);          // 3125 * 8 = 25000 waves
    gemm_bf16_2x2<<<waves * 32 / 256, 256, 0, stream>>>(xb, w1p, support1,
                                                        N_NODES, H1, F_IN);
    long long tot = (long long)N_EDGES * (H1 / 4);
    spmm_atomic<<<(unsigned)((tot + 255) / 256), 256, 0, stream>>>(support1, esrc, edst, ew,
                                                                   agg1, H1, 6, tot);
    long long te = (long long)N_NODES * H1;
    bn_relu_to_bf16<<<(unsigned)((te + 255) / 256), 256, 0, stream>>>(agg1, g1, b1, m1, v1,
                                                                      h1b, H1 - 1, te);
  }

  // Layer 2: WMMA GEMM -> SpMM -> fused BN2 + decoder + output writes.
  // (agg2/support2 alias xb, which is dead after the layer-1 GEMM above.)
  hipMemsetAsync(agg2, 0, (size_t)N_NODES * H2 * 4, stream);
  {
    int waves = (N_NODES / 32) * (H2 / 32);          // 3125 * 4 = 12500 waves
    gemm_bf16_2x2<<<(waves * 32 + 255) / 256, 256, 0, stream>>>(h1b, w2p, support2,
                                                                N_NODES, H2, H1);
    long long tot = (long long)N_EDGES * (H2 / 4);
    spmm_atomic<<<(unsigned)((tot + 255) / 256), 256, 0, stream>>>(support2, esrc, edst, ew,
                                                                   agg2, H2, 5, tot);
    finalize<<<(N_NODES * 32) / 256, 256, 0, stream>>>(agg2, g2, b2, m2, v2, linW, linb,
                                                       out, N_NODES);
  }
}
